// AbsorbQKVO_35914516529254
// MI455X (gfx1250) — compile-verified, matrix operand order checked by reading, not run on hardware
//
#include <hip/hip_runtime.h>

// ---------------------------------------------------------------------------
// MLA attention block for MI455X (gfx1250, wave32, WMMA 16x16x32 bf16).
// Compute-bound (~620 GFLOP vs ~15us of HBM traffic at 23.3 TB/s), so all
// GEMMs run as bf16 x bf16 -> f32 v_wmma_f32_16x16x32_bf16; softmax/RoPE/
// accumulation stay in f32. Dense GEMMs use the Tensor Data Mover to
// double-buffer 128x32 A/B tiles into LDS (TENSORcnt-tracked async DMA).
// ---------------------------------------------------------------------------

typedef __bf16 bf16;
typedef bf16  v16bf __attribute__((ext_vector_type(16)));
typedef bf16  v8bf  __attribute__((ext_vector_type(8)));
typedef float v8f   __attribute__((ext_vector_type(8)));
typedef unsigned int v4u __attribute__((ext_vector_type(4)));
typedef int   v8i __attribute__((ext_vector_type(8)));
typedef int   v4i __attribute__((ext_vector_type(4)));

union FragAB { v16bf v; v8bf h[2]; };

__device__ __forceinline__ v8f wmma_bf16(const FragAB& a, const FragAB& b, v8f c) {
  return __builtin_amdgcn_wmma_f32_16x16x32_bf16(
      /*neg_a=*/false, a.v, /*neg_b=*/false, b.v,
      /*c_mod=*/(short)0, c, /*reuse_a=*/false, /*reuse_b=*/false);
}

// Problem constants (match reference).
constexpr int PB   = 2;       // batch
constexpr int PS   = 2048;    // seq
constexpr int HID  = 2048;
constexpr int NH   = 16;      // heads
constexpr int QK_ROPE = 64;
constexpr int V_ROPE  = 64;
constexpr int KVR  = 512;
constexpr int QLR  = 1536;
constexpr int DQ   = QK_ROPE + KVR;          // 576
constexpr int QOUT = NH * DQ;                // 9216
constexpr int KVOUT = QK_ROPE + V_ROPE + KVR; // 640
constexpr int BS   = PB * PS;                // 4096 rows

// ---------------------------------------------------------------------------
// Tensor Data Mover: 2D tile load Global -> LDS.
// D# group0 (128b): [1:0]=count=1, [63:32]=lds byte addr,
//                   [120:64]=global byte addr, [127:126]=type=2.
// D# group1 (256b): [17:16]=data_size(1 => 2B), [79:48]=tensor_dim0,
//                   [111:80]=tensor_dim1, [127:112]=tile_dim0,
//                   [143:128]=tile_dim1, [207:160]=tensor_dim0_stride.
// Tile rows land consecutively in LDS (row-major [tile_d1][tile_d0]).
// All operands are workgroup-uniform -> SGPR descriptor groups.
// This toolchain exposes the 6-arg builtin: (g0, g1, g2, g3, g4, cpol).
// ---------------------------------------------------------------------------
__device__ __forceinline__ void tdm_load_2d(unsigned lds_byte_addr,
                                            const void* gptr,
                                            unsigned tile_d0, unsigned tile_d1,
                                            unsigned tensor_d0, unsigned tensor_d1,
                                            unsigned row_stride_elems) {
  const unsigned long long ga = (unsigned long long)(uintptr_t)gptr;
  v4u g0;
  g0[0] = 1u;                                            // count=1, user mode
  g0[1] = lds_byte_addr;
  g0[2] = (unsigned)ga;
  g0[3] = ((unsigned)(ga >> 32) & 0x01FFFFFFu) | 0x80000000u;  // type=2
  v8i g1;
  g1[0] = (int)(1u << 16);                               // data_size = 2 bytes
  g1[1] = (int)((tensor_d0 & 0xFFFFu) << 16);            // dim0[15:0]
  g1[2] = (int)(((tensor_d0 >> 16) & 0xFFFFu) | ((tensor_d1 & 0xFFFFu) << 16));
  g1[3] = (int)(((tensor_d1 >> 16) & 0xFFFFu) | ((tile_d0 & 0xFFFFu) << 16));
  g1[4] = (int)(tile_d1 & 0xFFFFu);                      // tile_dim1, tile_dim2=0
  g1[5] = (int)row_stride_elems;                         // dim0_stride[31:0]
  g1[6] = 0;                                             // stride[47:32], dim1_stride lo
  g1[7] = 0;
  const v4i z4 = {0, 0, 0, 0};
  const v8i z8 = {0, 0, 0, 0, 0, 0, 0, 0};
  __builtin_amdgcn_tensor_load_to_lds(g0, g1, z4, z4, z8, 0);
}

__device__ __forceinline__ unsigned lds_addr_of(const void* p) {
  // Generic LDS pointers carry the LDS offset in the low 32 bits.
  return (unsigned)(uintptr_t)p;
}

// ---------------------------------------------------------------------------
// f32 -> bf16 conversion (grid-stride, memory bound).
// ---------------------------------------------------------------------------
__global__ void cvt_f32_to_bf16(const float* __restrict__ src,
                                bf16* __restrict__ dst, size_t n) {
  size_t i = (size_t)blockIdx.x * blockDim.x + threadIdx.x;
  size_t stride = (size_t)gridDim.x * blockDim.x;
  for (; i < n; i += stride) dst[i] = (bf16)src[i];
}

// ---------------------------------------------------------------------------
// C(M,N) = A(M,K) * W(N,K)^T, bf16 in, WMMA f32 accum.
// Workgroup (8 waves) owns a 128x128 tile. Per 32-K step, wave 0 issues two
// TDM descriptors (A 128x32, W 128x32 -> LDS) for the *next* buffer, waits
// TENSORcnt<=2 for the current one, then all waves compute their 64x32
// sub-tile (4x2 fragments, 8 WMMAs) from LDS.
// Fragment layout (16-bit ISA striping): lane L holds row/col L&15;
// lanes 0-15 carry K {k0..k0+7, k0+16..k0+23}, lanes 16-31 carry
// K {k0+8..k0+15, k0+24..k0+31} -> two ds b128 loads per fragment.
// ---------------------------------------------------------------------------
template <bool OUT_F32>
__global__ __launch_bounds__(256) void gemm_bt_bf16(
    const bf16* __restrict__ A, const bf16* __restrict__ W,
    void* __restrict__ Cout, int M, int N, int K) {
  __shared__ __align__(16) bf16 sA[2][128 * 32];   // 8 KB x2
  __shared__ __align__(16) bf16 sB[2][128 * 32];   // 8 KB x2

  const int tid  = threadIdx.x;
  const int wave = tid >> 5, lane = tid & 31;
  const int l15  = lane & 15;
  const int hi   = lane >> 4;
  const int m0   = blockIdx.y * 128;
  const int n0   = blockIdx.x * 128;
  const int mw   = (wave & 1) * 64;    // wave M offset inside tile
  const int nw   = (wave >> 1) * 32;   // wave N offset inside tile

  v8f acc[4][2];
#pragma unroll
  for (int mf = 0; mf < 4; ++mf)
#pragma unroll
    for (int nf = 0; nf < 2; ++nf)
#pragma unroll
      for (int r = 0; r < 8; ++r) acc[mf][nf][r] = 0.0f;

  const int nk = K / 32;
  if (wave == 0) {
    tdm_load_2d(lds_addr_of(&sA[0][0]), A + (size_t)m0 * K, 32, 128,
                (unsigned)K, (unsigned)M, (unsigned)K);
    tdm_load_2d(lds_addr_of(&sB[0][0]), W + (size_t)n0 * K, 32, 128,
                (unsigned)K, (unsigned)N, (unsigned)K);
  }

  for (int i = 0; i < nk; ++i) {
    const int buf = i & 1;
    if (wave == 0) {
      if (i + 1 < nk) {
        const int k1 = (i + 1) * 32;
        tdm_load_2d(lds_addr_of(&sA[buf ^ 1][0]), A + (size_t)m0 * K + k1,
                    32, 128, (unsigned)K, (unsigned)M, (unsigned)K);
        tdm_load_2d(lds_addr_of(&sB[buf ^ 1][0]), W + (size_t)n0 * K + k1,
                    32, 128, (unsigned)K, (unsigned)N, (unsigned)K);
        __builtin_amdgcn_s_wait_tensorcnt(2);  // current buffer's 2 loads done
      } else {
        __builtin_amdgcn_s_wait_tensorcnt(0);
      }
    }
    __syncthreads();

    FragAB af[4], bfr[2];
#pragma unroll
    for (int mf = 0; mf < 4; ++mf) {
      const bf16* row = &sA[buf][(mw + mf * 16 + l15) * 32];
      af[mf].h[0] = *(const v8bf*)(row + 8 * hi);
      af[mf].h[1] = *(const v8bf*)(row + 16 + 8 * hi);
    }
#pragma unroll
    for (int nf = 0; nf < 2; ++nf) {
      const bf16* row = &sB[buf][(nw + nf * 16 + l15) * 32];
      bfr[nf].h[0] = *(const v8bf*)(row + 8 * hi);
      bfr[nf].h[1] = *(const v8bf*)(row + 16 + 8 * hi);
    }
#pragma unroll
    for (int mf = 0; mf < 4; ++mf)
#pragma unroll
      for (int nf = 0; nf < 2; ++nf)
        acc[mf][nf] = wmma_bf16(af[mf], bfr[nf], acc[mf][nf]);
    __syncthreads();
  }

  // C/D layout: lane holds col n=L&15; VGPR r holds row base + r + 8*hi.
#pragma unroll
  for (int mf = 0; mf < 4; ++mf) {
#pragma unroll
    for (int nf = 0; nf < 2; ++nf) {
      const int col = n0 + nw + nf * 16 + l15;
#pragma unroll
      for (int r = 0; r < 8; ++r) {
        const int row = m0 + mw + mf * 16 + r + 8 * hi;
        const float v = acc[mf][nf][r];
        if (OUT_F32) ((float*)Cout)[(size_t)row * N + col] = v;
        else         ((bf16*)Cout)[(size_t)row * N + col] = (bf16)v;
      }
    }
  }
}

// ---------------------------------------------------------------------------
// Build query (B,H,S,576) bf16 with RoPE on first 64 dims.
// q buffer is (B*S, 9216) bf16, per (s): head h at [h*576, h*576+576).
// ---------------------------------------------------------------------------
__global__ void build_query_kernel(const bf16* __restrict__ q,
                                   const float* __restrict__ cosp,
                                   const float* __restrict__ sinp,
                                   bf16* __restrict__ query) {
  const size_t total = (size_t)PB * NH * PS * DQ;
  size_t idx = (size_t)blockIdx.x * blockDim.x + threadIdx.x;
  size_t stride = (size_t)gridDim.x * blockDim.x;
  for (; idx < total; idx += stride) {
    const int d = (int)(idx % DQ);
    size_t t = idx / DQ;
    const int s = (int)(t % PS); t /= PS;
    const int h = (int)(t % NH);
    const int b = (int)(t / NH);
    const size_t base = ((size_t)b * PS + s) * QOUT + (size_t)h * DQ;
    float x = (float)q[base + d];
    if (d < QK_ROPE) {
      const size_t ci = ((size_t)b * PS + s) * QK_ROPE + d;
      const float cs = cosp[ci], sn = sinp[ci];
      const int dr = (d < 32) ? d + 32 : d - 32;
      const float other = (float)q[base + dr];
      const float rot = (d < 32) ? -other : other;
      x = x * cs + rot * sn;
    }
    query[idx] = (bf16)x;  // idx == ((b*NH+h)*PS+s)*DQ + d
  }
}

// ---------------------------------------------------------------------------
// Build key (B,S,576) = [RoPE(k_rope), kv_nope] and value (B,S,576) =
// [v_rope, kv_nope] from kv (B*S, 640).
// ---------------------------------------------------------------------------
__global__ void build_kv_kernel(const bf16* __restrict__ kv,
                                const float* __restrict__ cosp,
                                const float* __restrict__ sinp,
                                bf16* __restrict__ keyb,
                                bf16* __restrict__ valb) {
  const size_t total = (size_t)PB * PS * DQ;
  size_t idx = (size_t)blockIdx.x * blockDim.x + threadIdx.x;
  size_t stride = (size_t)gridDim.x * blockDim.x;
  for (; idx < total; idx += stride) {
    const int d = (int)(idx % DQ);
    size_t t = idx / DQ;
    const int s = (int)(t % PS);
    const int b = (int)(t / PS);
    const size_t base = ((size_t)b * PS + s) * KVOUT;
    float kx, vx;
    if (d < QK_ROPE) {
      const size_t ci = ((size_t)b * PS + s) * QK_ROPE + d;
      const float cs = cosp[ci], sn = sinp[ci];
      const float x = (float)kv[base + d];
      const int dr = (d < 32) ? d + 32 : d - 32;
      const float other = (float)kv[base + dr];
      kx = x * cs + ((d < 32) ? -other : other) * sn;
      vx = (float)kv[base + QK_ROPE + d];
    } else {
      const float nope = (float)kv[base + QK_ROPE + V_ROPE + (d - QK_ROPE)];
      kx = nope; vx = nope;
    }
    keyb[idx] = (bf16)kx;
    valb[idx] = (bf16)vx;
  }
}

// ---------------------------------------------------------------------------
// Flash attention. Block = 256 threads (8 waves) per (b, h, 32-query tile).
// Per 32-key chunk:
//   1) stage V^T chunk to LDS (so PV B-frags are contiguous ds b128 loads)
//      + waves 0-3 compute the 32x32 score tile (18 WMMA steps over d=576)
//   2) threads 0-31: online softmax update (f32, causal mask, v_exp_f32)
//   3) all 8 waves: rescale 9 f32 accum fragments (144-wide d-slice each)
//      and accumulate P(16x32) x V(32x144) with 9 WMMAs.
// Causality truncates the chunk loop. Output written bf16 to (B,S,9216).
// ---------------------------------------------------------------------------
#define AT_QT 32
#define AT_KC 32

__global__ __launch_bounds__(256) void attn_kernel(
    const bf16* __restrict__ Q,   // (B,H,S,576)
    const bf16* __restrict__ Kb_, // (B,S,576)
    const bf16* __restrict__ Vb_, // (B,S,576)
    bf16* __restrict__ O) {       // (B,S,9216)
  __shared__ __align__(16) bf16  sVT[DQ * AT_KC];     // [n][k]  36 KB
  __shared__ __align__(16) float sS[AT_QT * AT_KC];   // 4 KB
  __shared__ __align__(16) bf16  sP[AT_QT * AT_KC];   // 2 KB
  __shared__ float sM[AT_QT], sL[AT_QT], sA_[AT_QT];

  const int qt = blockIdx.x, h = blockIdx.y, b = blockIdx.z;
  const int q0 = qt * AT_QT;
  const int tid = threadIdx.x;
  const int wave = tid >> 5, lane = tid & 31;
  const int hi = lane >> 4, l15 = lane & 15;

  const bf16* Qb = Q + (((size_t)b * NH + h) * PS + q0) * DQ;
  const bf16* Kr = Kb_ + (size_t)b * PS * DQ;
  const bf16* Vr = Vb_ + (size_t)b * PS * DQ;

  if (tid < AT_QT) { sM[tid] = -3.0e38f; sL[tid] = 0.0f; }

  const int qsub = wave >> 2;          // 0..1 : 16-row query sub-tile
  const int dsl  = (wave & 3) * 144;   // d-slice base (9 fragments wide)
  v8f acc[9];
#pragma unroll
  for (int f = 0; f < 9; ++f)
#pragma unroll
    for (int r = 0; r < 8; ++r) acc[f][r] = 0.0f;

  const int nchunk = qt + 1;  // keys [0, q0+32) due to causal mask
  __syncthreads();

  for (int kc = 0; kc < nchunk; ++kc) {
    const int kbase = kc * AT_KC;

    // Phase 1a: transpose V chunk into LDS (coalesced global reads).
    for (int i = tid; i < DQ * AT_KC; i += 256) {
      const int n = i % DQ;
      const int k = i / DQ;
      sVT[n * AT_KC + k] = Vr[(size_t)(kbase + k) * DQ + n];
    }

    // Phase 1b: scores S = Q * K^T (waves 0-3 each own a 16x16 tile).
    if (wave < 4) {
      const int sq = wave >> 1, sk = wave & 1;
      const bf16* qrow = Qb + (size_t)(sq * 16 + l15) * DQ;
      const bf16* krow = Kr + (size_t)(kbase + sk * 16 + l15) * DQ;
      v8f s;
#pragma unroll
      for (int r = 0; r < 8; ++r) s[r] = 0.0f;
      for (int d0 = 0; d0 < DQ; d0 += 32) {
        FragAB a, bb;
        a.h[0]  = *(const v8bf*)(qrow + d0 + 8 * hi);
        a.h[1]  = *(const v8bf*)(qrow + d0 + 16 + 8 * hi);
        bb.h[0] = *(const v8bf*)(krow + d0 + 8 * hi);
        bb.h[1] = *(const v8bf*)(krow + d0 + 16 + 8 * hi);
        s = wmma_bf16(a, bb, s);
      }
#pragma unroll
      for (int r = 0; r < 8; ++r)
        sS[(sq * 16 + r + 8 * hi) * AT_KC + sk * 16 + l15] = s[r];
    }
    __syncthreads();

    // Phase 2: online softmax row update (f32).
    if (tid < AT_QT) {
      const float scale = 0.08838834764831845f;  // 1/sqrt(128)
      const int row = q0 + tid;
      const float mOld = sM[tid];
      float mNew = mOld;
      float vals[AT_KC];
#pragma unroll
      for (int c = 0; c < AT_KC; ++c) {
        float v = sS[tid * AT_KC + c] * scale;
        if (kbase + c > row) v += -1.0e9f;   // causal mask (matches ref)
        vals[c] = v;
        mNew = fmaxf(mNew, v);
      }
      const float alpha = __expf(mOld - mNew);
      float l = sL[tid] * alpha;
#pragma unroll
      for (int c = 0; c < AT_KC; ++c) {
        const float p = __expf(vals[c] - mNew);
        l += p;
        sP[tid * AT_KC + c] = (bf16)p;
      }
      sM[tid] = mNew; sL[tid] = l; sA_[tid] = alpha;
    }
    __syncthreads();

    // Phase 3: rescale accumulators and accumulate P x V (all 8 waves).
    {
      float scl[8];
#pragma unroll
      for (int r = 0; r < 8; ++r) scl[r] = sA_[qsub * 16 + r + 8 * hi];
#pragma unroll
      for (int f = 0; f < 9; ++f)
#pragma unroll
        for (int r = 0; r < 8; ++r) acc[f][r] *= scl[r];

      FragAB a;
      const bf16* prow = sP + (size_t)(qsub * 16 + l15) * AT_KC;
      a.h[0] = *(const v8bf*)(prow + 8 * hi);
      a.h[1] = *(const v8bf*)(prow + 16 + 8 * hi);
#pragma unroll
      for (int f = 0; f < 9; ++f) {
        const bf16* vtr = sVT + (size_t)(dsl + f * 16 + l15) * AT_KC;
        FragAB bb;
        bb.h[0] = *(const v8bf*)(vtr + 8 * hi);
        bb.h[1] = *(const v8bf*)(vtr + 16 + 8 * hi);
        acc[f] = wmma_bf16(a, bb, acc[f]);
      }
    }
    __syncthreads();
  }

  // Epilogue: divide by softmax sum, write (B,S,H*576) bf16.
  float linv[8];
#pragma unroll
  for (int r = 0; r < 8; ++r) linv[r] = 1.0f / sL[qsub * 16 + r + 8 * hi];
#pragma unroll
  for (int f = 0; f < 9; ++f) {
    const int col = h * DQ + dsl + f * 16 + l15;
#pragma unroll
    for (int r = 0; r < 8; ++r) {
      const int row = q0 + qsub * 16 + r + 8 * hi;
      O[((size_t)b * PS + row) * QOUT + col] = (bf16)(acc[f][r] * linv[r]);
    }
  }
}

// ---------------------------------------------------------------------------
// Host-side orchestration.
// ---------------------------------------------------------------------------
extern "C" void kernel_launch(void* const* d_in, const int* in_sizes, int n_in,
                              void* d_out, int out_size, void* d_ws, size_t ws_size,
                              hipStream_t stream) {
  (void)in_sizes; (void)n_in; (void)out_size; (void)ws_size;
  const float* hidden = (const float*)d_in[0];
  // d_in[1] = attention_mask (causal -1e9), computed analytically in-kernel.
  const float* cosp   = (const float*)d_in[2];
  const float* sinp   = (const float*)d_in[3];
  const float* w_qa   = (const float*)d_in[4];
  const float* w_qb   = (const float*)d_in[5];
  const float* w_kv   = (const float*)d_in[6];
  const float* w_o    = (const float*)d_in[7];
  float* out = (float*)d_out;

  // Bump-allocate bf16 scratch buffers from d_ws (~346 MB total).
  char* ws = (char*)d_ws;
  size_t off = 0;
  auto alloc_bf16 = [&](size_t elems) {
    bf16* p = (bf16*)(ws + off);
    off += elems * sizeof(bf16);
    return p;
  };
  bf16* hid_b  = alloc_bf16((size_t)BS * HID);       // 4096 x 2048
  bf16* wqa_b  = alloc_bf16((size_t)QLR * HID);      // 1536 x 2048
  bf16* wqb_b  = alloc_bf16((size_t)QOUT * QLR);     // 9216 x 1536
  bf16* wkv_b  = alloc_bf16((size_t)KVOUT * HID);    // 640 x 2048
  bf16* wo_b   = alloc_bf16((size_t)HID * QOUT);     // 2048 x 9216
  bf16* q_lat  = alloc_bf16((size_t)BS * QLR);       // 4096 x 1536
  bf16* qbuf   = alloc_bf16((size_t)BS * QOUT);      // 4096 x 9216
  bf16* kvbuf  = alloc_bf16((size_t)BS * KVOUT);     // 4096 x 640
  bf16* queryb = alloc_bf16((size_t)PB * NH * PS * DQ);
  bf16* keyb   = alloc_bf16((size_t)PB * PS * DQ);
  bf16* valb   = alloc_bf16((size_t)PB * PS * DQ);
  bf16* attn_o = alloc_bf16((size_t)BS * QOUT);      // 4096 x 9216

  const dim3 cvtGrid(2048), cvtBlk(256);
  cvt_f32_to_bf16<<<cvtGrid, cvtBlk, 0, stream>>>(hidden, hid_b, (size_t)BS * HID);
  cvt_f32_to_bf16<<<cvtGrid, cvtBlk, 0, stream>>>(w_qa, wqa_b, (size_t)QLR * HID);
  cvt_f32_to_bf16<<<cvtGrid, cvtBlk, 0, stream>>>(w_qb, wqb_b, (size_t)QOUT * QLR);
  cvt_f32_to_bf16<<<cvtGrid, cvtBlk, 0, stream>>>(w_kv, wkv_b, (size_t)KVOUT * HID);
  cvt_f32_to_bf16<<<cvtGrid, cvtBlk, 0, stream>>>(w_o, wo_b, (size_t)HID * QOUT);

  // q_lat = hidden @ w_qa^T  (4096,2048)x(1536,2048)^T
  gemm_bt_bf16<false><<<dim3(QLR / 128, BS / 128), 256, 0, stream>>>(
      hid_b, wqa_b, q_lat, BS, QLR, HID);
  // q = q_lat @ w_qb^T       (4096,1536)x(9216,1536)^T
  gemm_bt_bf16<false><<<dim3(QOUT / 128, BS / 128), 256, 0, stream>>>(
      q_lat, wqb_b, qbuf, BS, QOUT, QLR);
  // kv = hidden @ w_kv^T     (4096,2048)x(640,2048)^T
  gemm_bt_bf16<false><<<dim3(KVOUT / 128, BS / 128), 256, 0, stream>>>(
      hid_b, wkv_b, kvbuf, BS, KVOUT, HID);

  // RoPE + assemble query/key/value.
  build_query_kernel<<<dim3(4096), 256, 0, stream>>>(qbuf, cosp, sinp, queryb);
  build_kv_kernel<<<dim3(2048), 256, 0, stream>>>(kvbuf, cosp, sinp, keyb, valb);

  // Flash attention.
  attn_kernel<<<dim3(PS / AT_QT, NH, PB), 256, 0, stream>>>(
      queryb, keyb, valb, attn_o);

  // out = attn_o @ w_o^T     (4096,9216)x(2048,9216)^T -> f32 d_out
  gemm_bt_bf16<true><<<dim3(HID / 128, BS / 128), 256, 0, stream>>>(
      attn_o, wo_b, out, BS, HID, QOUT);
}
